// SO3Convolution_33036888441085
// MI455X (gfx1250) — matrix operand order repaired and checked by reading, not compiled, not run on hardware
//
#include <hip/hip_runtime.h>
#include <math.h>

// ---------------------------------------------------------------------------
// SO3 convolution for MI455X (gfx1250, wave32, WMMA).
// Strategy: per degree l (d = 2l+1) the einsum is a GEMM:
//   M = 2048*d (b,m), N = 128*d (y,v), K = 128*d (x,u)
// computed with v_wmma_f32_16x16x32_bf16 using a hi/lo bf16 split of both
// operands (3 WMMAs per product) for fp32-grade accuracy.
// ---------------------------------------------------------------------------

typedef __attribute__((ext_vector_type(16))) __bf16 bf16x16;
typedef __attribute__((ext_vector_type(8)))  __bf16 bf16x8;
typedef __attribute__((ext_vector_type(8)))  float  f32x8;

#define BATCH 2048
#define FCH   128   // f_in == f_out
#define KTOT  286   // sum (2l+1)^2, l=0..5
#define NGRID 168

__device__ __forceinline__ unsigned short f32_to_bf16_hi(float f) {
  unsigned int u = __float_as_uint(f);
  u += 0x7FFFu + ((u >> 16) & 1u);   // round-to-nearest-even on bf16 boundary
  return (unsigned short)(u >> 16);
}
__device__ __forceinline__ float bf16_bits_to_f32(unsigned short h) {
  return __uint_as_float(((unsigned int)h) << 16);
}

// ---------------------------------------------------------------------------
// Repack x[b][x][koff + u*d + m] -> A[(b*d+m)][(x*d+u)]  (bf16 hi + lo)
// Coalesced 2-byte writes; reads are short-strided gathers served by L0/L2.
// ---------------------------------------------------------------------------
template<int D>
__global__ void __launch_bounds__(256)
so3_repack_a(const float* __restrict__ xin,
             unsigned short* __restrict__ ahi,
             unsigned short* __restrict__ alo,
             int koff) {
  const int Kl = FCH * D;
  const long long total = (long long)BATCH * D * Kl;
  long long t = (long long)blockIdx.x * 256 + threadIdx.x;
  if (t >= total) return;
  int k   = (int)(t % Kl);
  int row = (int)(t / Kl);
  int b = row / D, m = row % D;
  int x = k   / D, u = k   % D;
  float val = xin[((size_t)b * FCH + x) * KTOT + koff + u * D + m];
  unsigned short hi = f32_to_bf16_hi(val);
  ahi[t] = hi;
  alo[t] = f32_to_bf16_hi(val - bf16_bits_to_f32(hi));
}

// ---------------------------------------------------------------------------
// Fuse psi synthesis + normalization + transpose:
//   W[(y*d+v)][(x*d+u)] = scale * sum_g D[g][koff+u*d+v] * w[x][y][g]
// One block per (x,y); w-row staged in LDS; one thread per (u,v).
// ---------------------------------------------------------------------------
template<int D>
__global__ void __launch_bounds__(128)
so3_build_w(const float* __restrict__ wg,
            const float* __restrict__ Dmat,
            unsigned short* __restrict__ whi,
            unsigned short* __restrict__ wlo,
            int koff, float scale) {
  const int Kl = FCH * D;
  __shared__ float wrow[NGRID];
  const int blk = blockIdx.x;          // x*FCH + y
  const int x = blk >> 7;
  const int y = blk & (FCH - 1);
  for (int g = threadIdx.x; g < NGRID; g += 128)
    wrow[g] = wg[(size_t)blk * NGRID + g];
  __syncthreads();
  const int j = threadIdx.x;
  if (j < D * D) {
    const int u = j / D, v = j % D;
    const int di = koff + u * D + v;
    float s = 0.f;
    #pragma unroll 4
    for (int g = 0; g < NGRID; ++g)
      s += Dmat[(size_t)g * KTOT + di] * wrow[g];
    s *= scale;
    unsigned short hi = f32_to_bf16_hi(s);
    size_t idx = (size_t)(y * D + v) * Kl + (size_t)(x * D + u);
    whi[idx] = hi;
    wlo[idx] = f32_to_bf16_hi(s - bf16_bits_to_f32(hi));
  }
}

// ---------------------------------------------------------------------------
// WMMA GEMM. p points at (row base + k + sel-dependent offset); second half of
// the 16-element fragment sits gapElems further along (16 for both A and B).
// ---------------------------------------------------------------------------
__device__ __forceinline__ bf16x16 load_frag(const unsigned short* p, int gapElems) {
  bf16x8 a = *(const bf16x8*)(p);
  bf16x8 b = *(const bf16x8*)(p + gapElems);
  return __builtin_shufflevector(a, b, 0,1,2,3,4,5,6,7,8,9,10,11,12,13,14,15);
}

#define WMMA_BF16(A_, B_, C_) \
  __builtin_amdgcn_wmma_f32_16x16x32_bf16(false, (A_), false, (B_), (short)0, (C_), false, false)

template<int D>
__global__ void __launch_bounds__(256)
so3_wmma_gemm(const unsigned short* __restrict__ ahi,
              const unsigned short* __restrict__ alo,
              const unsigned short* __restrict__ whi,
              const unsigned short* __restrict__ wlo,
              float* __restrict__ out, int koff) {
  const int Kl  = FCH * D;             // K dim, multiple of 32
  const int lane = threadIdx.x & 31;
  const int wave = threadIdx.x >> 5;   // 8 waves
  const int l15  = lane & 15;
  const int sel  = lane >> 4;
  const int r0 = blockIdx.y * 128 + (wave & 3) * 32;   // wave tile 32(M) x 64(N)
  const int c0 = blockIdx.x * 128 + (wave >> 2) * 64;

  f32x8 zero = {};
  f32x8 acc[2][4];
  #pragma unroll
  for (int mi = 0; mi < 2; ++mi)
    #pragma unroll
    for (int ni = 0; ni < 4; ++ni)
      acc[mi][ni] = zero;

  // A fragment bases: rows r0+l15 and r0+16+l15, K-offset sel*8 (ISA layout)
  const unsigned short* pa0h = ahi + (size_t)(r0 +      l15) * Kl + sel * 8;
  const unsigned short* pa1h = ahi + (size_t)(r0 + 16 + l15) * Kl + sel * 8;
  const unsigned short* pa0l = alo + (size_t)(r0 +      l15) * Kl + sel * 8;
  const unsigned short* pa1l = alo + (size_t)(r0 + 16 + l15) * Kl + sel * 8;
  // B fragment bases: column c0+ni*16+l15, K-offset sel*8 (ISA layout)
  const unsigned short* pbh[4];
  const unsigned short* pbl[4];
  #pragma unroll
  for (int ni = 0; ni < 4; ++ni) {
    pbh[ni] = whi + (size_t)(c0 + ni * 16 + l15) * Kl + sel * 8;
    pbl[ni] = wlo + (size_t)(c0 + ni * 16 + l15) * Kl + sel * 8;
  }

  for (int k0 = 0; k0 < Kl; k0 += 32) {
    bf16x16 a0h = load_frag(pa0h + k0, 16);
    bf16x16 a1h = load_frag(pa1h + k0, 16);
    bf16x16 a0l = load_frag(pa0l + k0, 16);
    bf16x16 a1l = load_frag(pa1l + k0, 16);
    #pragma unroll
    for (int ni = 0; ni < 4; ++ni) {
      bf16x16 bh = load_frag(pbh[ni] + k0, 16);
      bf16x16 bl = load_frag(pbl[ni] + k0, 16);
      // (ahi+alo)*(bhi+blo) ~= ahi*bhi + alo*bhi + ahi*blo   (drop lo*lo)
      acc[0][ni] = WMMA_BF16(a0h, bh, acc[0][ni]);
      acc[1][ni] = WMMA_BF16(a1h, bh, acc[1][ni]);
      acc[0][ni] = WMMA_BF16(a0l, bh, acc[0][ni]);
      acc[1][ni] = WMMA_BF16(a1l, bh, acc[1][ni]);
      acc[0][ni] = WMMA_BF16(a0h, bl, acc[0][ni]);
      acc[1][ni] = WMMA_BF16(a1h, bl, acc[1][ni]);
    }
  }

  // Scatter epilogue back to reference layout out[b][y][koff + v*d + m].
  // C layout: VGPR i -> row r0+mi*16+sel*8+i, col c0+ni*16+l15.
  #pragma unroll
  for (int mi = 0; mi < 2; ++mi)
    #pragma unroll
    for (int ni = 0; ni < 4; ++ni)
      #pragma unroll
      for (int i = 0; i < 8; ++i) {
        int r = r0 + mi * 16 + sel * 8 + i;
        int c = c0 + ni * 16 + l15;
        int bb = r / D, mm = r % D;
        int yy = c / D, vv = c % D;
        out[((size_t)(bb * FCH + yy)) * KTOT + koff + vv * D + mm] = acc[mi][ni][i];
      }
}

// ---------------------------------------------------------------------------
// Host side
// ---------------------------------------------------------------------------
template<int D>
static void run_degree(const float* x, const float* w, const float* Dmat, float* out,
                       unsigned short* Ahi, unsigned short* Alo,
                       unsigned short* Whi, unsigned short* Wlo,
                       int koff, hipStream_t stream) {
  const int Kl = FCH * D;
  const long long aTot = (long long)BATCH * D * Kl;
  int aBlocks = (int)((aTot + 255) / 256);
  so3_repack_a<D><<<aBlocks, 256, 0, stream>>>(x, Ahi, Alo, koff);
  float scale = 1.0f / (sqrtf((float)NGRID) * sqrtf((float)(FCH * D)));
  so3_build_w<D><<<FCH * FCH, 128, 0, stream>>>(w, Dmat, Whi, Wlo, koff, scale);
  dim3 grid((FCH * D) / 128, (BATCH * D) / 128);   // N, M tiles (exact multiples)
  so3_wmma_gemm<D><<<grid, 256, 0, stream>>>(Ahi, Alo, Whi, Wlo, out, koff);
}

extern "C" void kernel_launch(void* const* d_in, const int* in_sizes, int n_in,
                              void* d_out, int out_size, void* d_ws, size_t ws_size,
                              hipStream_t stream) {
  (void)in_sizes; (void)n_in; (void)out_size; (void)ws_size;
  const float* x    = (const float*)d_in[0];   // (2048, 128, 286)
  const float* w    = (const float*)d_in[1];   // (128, 128, 168)
  const float* Dmat = (const float*)d_in[2];   // (168, 286)
  float* out = (float*)d_out;                  // (2048, 128, 286)

  // Workspace (reused across degrees; sized for d=11):
  //   A: 2048*11 x 128*11 bf16 (hi+lo) = 2*31,719,424 ushorts
  //   W:  128*11 x 128*11 bf16 (hi+lo) = 2* 1,982,464 ushorts
  // total ~135 MB
  const size_t A_MAX = (size_t)BATCH * 11 * (size_t)FCH * 11;
  const size_t W_MAX = (size_t)FCH  * 11 * (size_t)FCH * 11;
  unsigned short* Ahi = (unsigned short*)d_ws;
  unsigned short* Alo = Ahi + A_MAX;
  unsigned short* Whi = Alo + A_MAX;
  unsigned short* Wlo = Whi + W_MAX;

  run_degree<1>(x, w, Dmat, out, Ahi, Alo, Whi, Wlo,   0, stream);
  run_degree<3 >(x, w, Dmat, out, Ahi, Alo, Whi, Wlo,   1, stream);
  run_degree<5 >(x, w, Dmat, out, Ahi, Alo, Whi, Wlo,  10, stream);
  run_degree<7 >(x, w, Dmat, out, Ahi, Alo, Whi, Wlo,  35, stream);
  run_degree<9 >(x, w, Dmat, out, Ahi, Alo, Whi, Wlo,  84, stream);
  run_degree<11>(x, w, Dmat, out, Ahi, Alo, Whi, Wlo, 165, stream);
}
